// SelfAttention_23983097381469
// MI455X (gfx1250) — compile-verified, matrix operand order checked by reading, not run on hardware
//
#include <hip/hip_runtime.h>
#include <hip/hip_bf16.h>

#define N_HEAD 12
#define BB 8
#define TT 1024
#define CC 768
#define DD 64
#define C3 (3 * CC)
#define KSTEPS (CC / 32)   // 24

typedef __bf16 bf16_t;
typedef __attribute__((ext_vector_type(16))) __bf16 v16bf;
typedef __attribute__((ext_vector_type(8)))  __bf16 v8bf;
typedef __attribute__((ext_vector_type(8)))  float   v8f;

// ---------------------------------------------------------------------------
// Fragment loader (bf16, 16x32 A-fragment / 32x16 B-fragment, wave32).
// ISA layout (CDNA5 ISA, 16-bit A 16x32):
//   lane L: r = L%16, kb = (L<16 ? 0 : 8)
//   elems 0..7  = src[r][kb+0 .. kb+7]
//   elems 8..15 = src[r][kb+16 .. kb+23]
// Works for global or LDS source (generic pointer); all 16B aligned.
// ---------------------------------------------------------------------------
static __device__ __forceinline__ v16bf load_frag(const bf16_t* base, int ld) {
    const int lane = threadIdx.x & 31;
    const int r    = lane & 15;
    const int kb   = (lane >> 4) << 3;   // 0 or 8
    const bf16_t* p = base + r * ld + kb;
    v8bf lo = *(const v8bf*)(p);
    v8bf hi = *(const v8bf*)(p + 16);
    v16bf out;
#pragma unroll
    for (int i = 0; i < 8; ++i) { out[i] = lo[i]; out[i + 8] = hi[i]; }
    return out;
}

static __device__ __forceinline__ v8f wmma_bf16(v16bf a, v16bf b, v8f c) {
    return __builtin_amdgcn_wmma_f32_16x16x32_bf16(
        /*neg_a=*/false, a, /*neg_b=*/false, b,
        /*c_mod=*/(short)0, c, /*reuse_a=*/false, /*reuse_b=*/false);
}

// ---------------------------------------------------------------------------
// CDNA5 async copy global -> LDS (ASYNCcnt-tracked, no VGPR staging).
// Inline asm per CDNA5_HIP.md recommendation (portable across toolchains).
// lds_byte = low 32 bits of generic pointer to __shared__ (== LDS offset).
// ---------------------------------------------------------------------------
static __device__ __forceinline__ void async_copy_b128(uint32_t lds_byte, const void* gaddr) {
    asm volatile("global_load_async_to_lds_b128 %0, %1, off"
                 :: "v"(lds_byte), "v"((unsigned long long)(uintptr_t)gaddr)
                 : "memory");
}
static __device__ __forceinline__ void wait_async0() {
    asm volatile("s_wait_asynccnt 0x0" ::: "memory");
}

// ---------------------------------------------------------------------------
// Pre-pass conversions
// ---------------------------------------------------------------------------
__global__ void k_cvt(const float* __restrict__ src, bf16_t* __restrict__ dst, int n) {
    int i = blockIdx.x * blockDim.x + threadIdx.x;
    if (i < n) dst[i] = (bf16_t)src[i];
}

// dst[n * rows + k] = src[k * cols + n]   (rows = K, cols = N)
__global__ void k_cvt_t(const float* __restrict__ src, bf16_t* __restrict__ dst,
                        int rows, int cols) {
    int i = blockIdx.x * blockDim.x + threadIdx.x;
    if (i >= rows * cols) return;
    int n = i / rows;
    int k = i - n * rows;
    dst[i] = (bf16_t)src[k * cols + n];
}

// ---------------------------------------------------------------------------
// Async staging of a 64x32 bf16 B tile (4 KB) into LDS.
// 128 threads x 2 b128 chunks. Tile row c = output col n0+c, ld = 32.
// ---------------------------------------------------------------------------
static __device__ __forceinline__ void issue_btile(bf16_t* dst /*LDS, 64*32*/,
                                                   const bf16_t* __restrict__ wt,
                                                   int n0, int ks) {
    const int e0 = threadIdx.x * 2;
#pragma unroll
    for (int u = 0; u < 2; ++u) {
        int e = e0 + u;
        int c = e >> 2, piece = e & 3;
        async_copy_b128((uint32_t)(uintptr_t)(dst + c * 32 + piece * 8),
                        wt + (size_t)(n0 + c) * CC + ks + piece * 8);
    }
}

// ---------------------------------------------------------------------------
// QKV GEMM: [8192,768] x [768,2304] + bias; scatter into q/k (B,H,T,D) and
// v transposed (B,H,D,T). One wave: 16x64 tile; block (4 waves): 64x64 tile.
// B tile shared through LDS via async copies, ping-pong double buffered:
//   iter j: s_wait_asynccnt 0 (own chunks of tile j) -> barrier (all chunks)
//           -> issue tile j+1 into other buffer -> WMMAs on tile j.
// Issue into buffer b happens only after the barrier that follows every
// wave's reads of b. Fully unrolled (KSTEPS compile-time) so accumulators
// stay pinned: no loop-carried PHI copies / hazard NOPs between WMMAs.
// ---------------------------------------------------------------------------
__global__ void __launch_bounds__(128) k_qkv(const bf16_t* __restrict__ xb,
                                             const bf16_t* __restrict__ wt,   // [2304][768]
                                             const float*  __restrict__ bias, // [2304]
                                             bf16_t* __restrict__ qws,
                                             bf16_t* __restrict__ kws,
                                             bf16_t* __restrict__ vtws) {
    __shared__ __align__(16) bf16_t bsh[2][64 * 32];
    const int wid  = threadIdx.x >> 5;
    const int lane = threadIdx.x & 31;
    const int m0 = blockIdx.y * 64 + wid * 16;
    const int n0 = blockIdx.x * 64;

    issue_btile(&bsh[0][0], wt, n0, 0);

    v8f acc[4] = {};
#pragma unroll
    for (int j = 0; j < KSTEPS; ++j) {
        const int buf = j & 1;
        wait_async0();
        __syncthreads();
        if (j + 1 < KSTEPS)
            issue_btile(&bsh[buf ^ 1][0], wt, n0, (j + 1) * 32);

        v16bf a = load_frag(xb + (size_t)m0 * CC + j * 32, CC);
#pragma unroll
        for (int dt = 0; dt < 4; ++dt) {
            v16bf bf = load_frag(&bsh[buf][dt * 16 * 32], 32);
            acc[dt] = wmma_bf16(a, bf, acc[dt]);
        }
    }

    const int n   = lane & 15;
    const int hb8 = (lane >> 4) << 3;
#pragma unroll
    for (int dt = 0; dt < 4; ++dt) {
#pragma unroll
        for (int i = 0; i < 8; ++i) {
            int r  = m0 + i + hb8;          // row in [0,8192)
            int c3 = n0 + dt * 16 + n;      // col in [0,2304)
            float val = acc[dt][i] + bias[c3];
            int b = r >> 10, t = r & 1023;
            int sec = c3 / CC, c = c3 - sec * CC;
            int h = c >> 6, d = c & 63;
            int bh = b * N_HEAD + h;
            if (sec == 0)      qws[((size_t)bh * TT + t) * DD + d]  = (bf16_t)val;
            else if (sec == 1) kws[((size_t)bh * TT + t) * DD + d]  = (bf16_t)val;
            else               vtws[((size_t)bh * DD + d) * TT + t] = (bf16_t)val;
        }
    }
}

// ---------------------------------------------------------------------------
// Flash attention: one wave per (b,h, 16 query rows). 32 keys per step:
// 4 WMMAs for scores, causal mask + online softmax (f32), P staged via LDS
// to A-fragment layout, 4 WMMAs for P*V into 16x64 f32 accumulator.
// ---------------------------------------------------------------------------
__global__ void __launch_bounds__(128) k_attn(const bf16_t* __restrict__ qws,
                                              const bf16_t* __restrict__ kws,
                                              const bf16_t* __restrict__ vtws,
                                              bf16_t* __restrict__ yb) {
    __shared__ __align__(16) bf16_t psh[4][16 * 32];
    const int wid  = threadIdx.x >> 5;
    const int lane = threadIdx.x & 31;
    const int gw = blockIdx.x * 4 + wid;        // 6144 waves total
    const int bh = gw >> 6;                     // (b*H + h)
    const int q0 = (gw & 63) * 16;

    const bf16_t* qbase = qws  + ((size_t)bh * TT + q0) * DD;
    const bf16_t* kbase = kws  + (size_t)bh * TT * DD;
    const bf16_t* vbase = vtws + (size_t)bh * DD * TT;

    v16bf qa0 = load_frag(qbase + 0,  DD);
    v16bf qa1 = load_frag(qbase + 32, DD);

    v8f o[4] = {};
    float mrow[8], lrow[8];
#pragma unroll
    for (int i = 0; i < 8; ++i) { mrow[i] = -3.0e38f; lrow[i] = 0.f; }

    const int n   = lane & 15;
    const int hb8 = (lane >> 4) << 3;
    bf16_t* pw = &psh[wid][0];

    for (int kt = 0; kt <= q0 + 15; kt += 32) {   // uniform per wave
        v8f s0 = {}, s1 = {};
        {
            v16bf kb0 = load_frag(kbase + (size_t)kt * DD + 0,  DD);
            v16bf kb1 = load_frag(kbase + (size_t)kt * DD + 32, DD);
            s0 = wmma_bf16(qa0, kb0, s0);
            s0 = wmma_bf16(qa1, kb1, s0);
            v16bf kb2 = load_frag(kbase + (size_t)(kt + 16) * DD + 0,  DD);
            v16bf kb3 = load_frag(kbase + (size_t)(kt + 16) * DD + 32, DD);
            s1 = wmma_bf16(qa0, kb2, s1);
            s1 = wmma_bf16(qa1, kb3, s1);
        }
#pragma unroll
        for (int i = 0; i < 8; ++i) {
            int m = i + hb8;
            int qidx = q0 + m;
            float v0 = s0[i] * 0.125f;           // 1/sqrt(64)
            float v1 = s1[i] * 0.125f;
            if (kt + n      > qidx) v0 = -3.0e38f;
            if (kt + 16 + n > qidx) v1 = -3.0e38f;
            float rmax = fmaxf(v0, v1);
#pragma unroll
            for (int off = 1; off < 16; off <<= 1)
                rmax = fmaxf(rmax, __shfl_xor(rmax, off, 32));
            float mnew = fmaxf(mrow[i], rmax);
            float corr = __expf(mrow[i] - mnew);
            float p0 = __expf(v0 - mnew);
            float p1 = __expf(v1 - mnew);
            float rs = p0 + p1;
#pragma unroll
            for (int off = 1; off < 16; off <<= 1)
                rs += __shfl_xor(rs, off, 32);
            lrow[i] = lrow[i] * corr + rs;
            mrow[i] = mnew;
#pragma unroll
            for (int dt = 0; dt < 4; ++dt) o[dt][i] *= corr;
            pw[m * 32 + n]      = (bf16_t)p0;    // P tile, row-major 16x32
            pw[m * 32 + 16 + n] = (bf16_t)p1;
        }
        v16bf pa = load_frag(pw, 32);            // per-wave LDS; DScnt ordered
#pragma unroll
        for (int dt = 0; dt < 4; ++dt) {
            v16bf vb = load_frag(vbase + (size_t)(dt * 16) * TT + kt, TT);
            o[dt] = wmma_bf16(pa, vb, o[dt]);
        }
    }

#pragma unroll
    for (int i = 0; i < 8; ++i) {
        float inv = 1.0f / lrow[i];
#pragma unroll
        for (int dt = 0; dt < 4; ++dt) o[dt][i] *= inv;
    }

    const int bt = (bh / N_HEAD) * TT;
    const int hc = (bh % N_HEAD) * DD;
#pragma unroll
    for (int dt = 0; dt < 4; ++dt)
#pragma unroll
        for (int i = 0; i < 8; ++i) {
            int m = i + hb8;
            yb[(size_t)(bt + q0 + m) * CC + hc + dt * 16 + n] = (bf16_t)o[dt][i];
        }
}

// ---------------------------------------------------------------------------
// Output projection: [8192,768] x [768,768] + bias -> f32 out.
// Same fully-unrolled async-LDS double-buffered B pipeline as k_qkv.
// ---------------------------------------------------------------------------
__global__ void __launch_bounds__(128) k_proj(const bf16_t* __restrict__ yb,
                                              const bf16_t* __restrict__ wt,   // [768][768]
                                              const float*  __restrict__ bias,
                                              float* __restrict__ out) {
    __shared__ __align__(16) bf16_t bsh[2][64 * 32];
    const int wid  = threadIdx.x >> 5;
    const int lane = threadIdx.x & 31;
    const int m0 = blockIdx.y * 64 + wid * 16;
    const int n0 = blockIdx.x * 64;

    issue_btile(&bsh[0][0], wt, n0, 0);

    v8f acc[4] = {};
#pragma unroll
    for (int j = 0; j < KSTEPS; ++j) {
        const int buf = j & 1;
        wait_async0();
        __syncthreads();
        if (j + 1 < KSTEPS)
            issue_btile(&bsh[buf ^ 1][0], wt, n0, (j + 1) * 32);

        v16bf a = load_frag(yb + (size_t)m0 * CC + j * 32, CC);
#pragma unroll
        for (int dt = 0; dt < 4; ++dt) {
            v16bf bf = load_frag(&bsh[buf][dt * 16 * 32], 32);
            acc[dt] = wmma_bf16(a, bf, acc[dt]);
        }
    }

    const int n   = lane & 15;
    const int hb8 = (lane >> 4) << 3;
#pragma unroll
    for (int dt = 0; dt < 4; ++dt)
#pragma unroll
        for (int i = 0; i < 8; ++i) {
            int r = m0 + i + hb8;
            int c = n0 + dt * 16 + n;
            out[(size_t)r * CC + c] = acc[dt][i] + bias[c];
        }
}

// ---------------------------------------------------------------------------
extern "C" void kernel_launch(void* const* d_in, const int* in_sizes, int n_in,
                              void* d_out, int out_size, void* d_ws, size_t ws_size,
                              hipStream_t stream) {
    const float* x      = (const float*)d_in[0];
    const float* W_attn = (const float*)d_in[1];
    const float* b_attn = (const float*)d_in[2];
    const float* W_proj = (const float*)d_in[3];
    const float* b_proj = (const float*)d_in[4];
    float* out = (float*)d_out;

    char* ws = (char*)d_ws;
    size_t off = 0;
    auto carve = [&](size_t bytes) -> char* {
        char* p = ws + off;
        off += (bytes + 255) & ~(size_t)255;
        return p;
    };
    const size_t M = (size_t)BB * TT;            // 8192
    bf16_t* xb   = (bf16_t*)carve(M * CC * 2);               // x in bf16
    bf16_t* wta  = (bf16_t*)carve((size_t)C3 * CC * 2);      // W_attn^T bf16 [2304][768]
    bf16_t* wtp  = (bf16_t*)carve((size_t)CC * CC * 2);      // W_proj^T bf16 [768][768]
    bf16_t* qws  = (bf16_t*)carve((size_t)BB * N_HEAD * TT * DD * 2);
    bf16_t* kws  = (bf16_t*)carve((size_t)BB * N_HEAD * TT * DD * 2);
    bf16_t* vtws = (bf16_t*)carve((size_t)BB * N_HEAD * DD * TT * 2);
    bf16_t* ybuf = (bf16_t*)carve(M * CC * 2);               // attention output bf16

    // 1) conversions
    {
        int n = (int)(M * CC);
        k_cvt<<<n / 256, 256, 0, stream>>>(x, xb, n);
        int na = C3 * CC;
        k_cvt_t<<<(na + 255) / 256, 256, 0, stream>>>(W_attn, wta, CC, C3);
        int np = CC * CC;
        k_cvt_t<<<(np + 255) / 256, 256, 0, stream>>>(W_proj, wtp, CC, CC);
    }
    // 2) QKV projection
    {
        dim3 grid(C3 / 64, (unsigned)(M / 64));   // 36 x 128
        k_qkv<<<grid, 128, 0, stream>>>(xb, wta, b_attn, qws, kws, vtws);
    }
    // 3) flash attention (6144 waves, 4 waves/block)
    {
        int waves = BB * N_HEAD * (TT / 16);
        k_attn<<<waves / 4, 128, 0, stream>>>(qws, kws, vtws, ybuf);
    }
    // 4) output projection
    {
        dim3 grid(CC / 64, (unsigned)(M / 64));   // 12 x 128
        k_proj<<<grid, 128, 0, stream>>>(ybuf, wtp, b_proj, out);
    }
}
